// SpatiotemporalGNN_37761352466984
// MI455X (gfx1250) — matrix operand order repaired and checked by reading, not compiled, not run on hardware
//
#include <hip/hip_runtime.h>
#include <hip/hip_bf16.h>
#include <math.h>
#include <stddef.h>

#define B_   32
#define E_   3
#define N_   30
#define S_   2048
#define FT   6144          // E_*S_
#define HID  16
#define TT   24            // interior timesteps per workgroup tile
#define TJ   (TT + 2)      // 26: mid-stage range (one conv halo consumed)
#define TX   (TT + 4)      // 28: input range (two conv halos)
#define NTILES (FT / TT)   // 256
#define EPSV 1e-5f

typedef __attribute__((ext_vector_type(16))) __bf16          v16bf;
typedef __attribute__((ext_vector_type(16))) unsigned short  v16us;
typedef __attribute__((ext_vector_type(8)))  float           v8f;

static __device__ __forceinline__ unsigned short f2bf(float f) {
  unsigned u = __builtin_bit_cast(unsigned, f);
  u += 0x7FFFu + ((u >> 16) & 1u);                // round-to-nearest-even
  return (unsigned short)(u >> 16);
}
static __device__ __forceinline__ float bf2f(unsigned short s) {
  unsigned u = ((unsigned)s) << 16;
  return __builtin_bit_cast(float, u);
}
static __device__ __forceinline__ float sigm(float x) { return 1.f / (1.f + __expf(-x)); }

static __device__ __forceinline__ v8f wmma_bf16(v16us a, v16us b, v8f c) {
  return __builtin_amdgcn_wmma_f32_16x16x32_bf16(
      false, __builtin_bit_cast(v16bf, a),
      false, __builtin_bit_cast(v16bf, b),
      (short)0, c, false, false);
}

// CDNA5 16-bit A fragment (16x32): lane holds row M=lane&15; VGPR v holds K pair.
static __device__ __forceinline__ int a_kb(int v, int lane) {
  return (v < 4 ? 2 * v : 16 + 2 * (v - 4)) + ((lane & 16) ? 8 : 0);
}
// CDNA5 16-bit B fragment (32x16): lane holds col N=lane&15; VGPR r holds K pair.
static __device__ __forceinline__ int b_kb(int r, int lane) {
  return (r < 4 ? 2 * r : 16 + 2 * (r - 4)) + ((lane & 16) ? 8 : 0);
}
// f32 C/D fragment (16x16): VGPR r -> row M = r + (lane>=16 ? 8 : 0), col N = lane&15.
static __device__ __forceinline__ int c_row(int r, int lane) {
  return r + ((lane & 16) ? 8 : 0);
}

// ---------------------------------------------------------------------------
// Chebyshev graph conv (order 3), fully in LDS, bf16 WMMA.
//   tc = lt @ h1 ; tn = (2*lt) @ tc - h1 ; h2 = relu([h1|tc]@[w0;w1] + tn@[w2;0] + b)
// LDS tensors laid out [32 nodes][TJ t][16 ch] as ushort(bf16); lt padded 30->32.
// ---------------------------------------------------------------------------
static __device__ __forceinline__ void cheb_stage(
    unsigned short* h1, unsigned short* tc, unsigned short* tn, unsigned short* h2,
    const unsigned short* ltb, const unsigned short* lt2b,
    const float* __restrict__ chw, const float* __restrict__ chb,
    int lane, int wave) {
  const int col = lane & 15;
  // pass A: tc = lt @ h1   (M = node tile, K = 32 nodes, N = 16 ch, one t per WMMA)
  for (int task = wave; task < TJ * 2; task += 8) {
    int j = task >> 1, nt = task & 1;
    v16us a, bb; v8f acc = {};
    int mrow = nt * 16 + (lane & 15);
#pragma unroll
    for (int v = 0; v < 8; ++v) {
      int kb = a_kb(v, lane);
      a[2 * v]     = ltb[mrow * 32 + kb];
      a[2 * v + 1] = ltb[mrow * 32 + kb + 1];
    }
#pragma unroll
    for (int r = 0; r < 8; ++r) {
      int kb = b_kb(r, lane);
      bb[2 * r]     = h1[(kb * TJ + j) * 16 + col];
      bb[2 * r + 1] = h1[((kb + 1) * TJ + j) * 16 + col];
    }
    acc = wmma_bf16(a, bb, acc);
#pragma unroll
    for (int r = 0; r < 8; ++r)
      tc[((nt * 16 + c_row(r, lane)) * TJ + j) * 16 + col] = f2bf(acc[r]);
  }
  __syncthreads();
  // pass B: tn = 2*lt @ tc - h1   (C preloaded with -h1)
  for (int task = wave; task < TJ * 2; task += 8) {
    int j = task >> 1, nt = task & 1;
    v16us a, bb; v8f acc;
    int mrow = nt * 16 + (lane & 15);
#pragma unroll
    for (int v = 0; v < 8; ++v) {
      int kb = a_kb(v, lane);
      a[2 * v]     = lt2b[mrow * 32 + kb];
      a[2 * v + 1] = lt2b[mrow * 32 + kb + 1];
    }
#pragma unroll
    for (int r = 0; r < 8; ++r) {
      int kb = b_kb(r, lane);
      bb[2 * r]     = tc[(kb * TJ + j) * 16 + col];
      bb[2 * r + 1] = tc[((kb + 1) * TJ + j) * 16 + col];
    }
#pragma unroll
    for (int r = 0; r < 8; ++r)
      acc[r] = -bf2f(h1[((nt * 16 + c_row(r, lane)) * TJ + j) * 16 + col]);
    acc = wmma_bf16(a, bb, acc);
#pragma unroll
    for (int r = 0; r < 8; ++r)
      tn[((nt * 16 + c_row(r, lane)) * TJ + j) * 16 + col] = f2bf(acc[r]);
  }
  __syncthreads();
  // pass C: channel mixing, w0/w1 packed into one K=32 WMMA, w2 in second.
  for (int task = wave; task < N_ * 2; task += 8) {
    int n = task >> 1, tb = task & 1;
    int trow = tb * 16 + (lane & 15);
    bool vld = trow < TJ;
    v16us a01, a2, bw01, bw2;
#pragma unroll
    for (int v = 0; v < 8; ++v) {
      int kb = a_kb(v, lane);
#pragma unroll
      for (int e = 0; e < 2; ++e) {
        int k = kb + e;
        unsigned short e01 = 0, e2 = 0;
        if (vld) {
          e01 = (k < 16) ? h1[(n * TJ + trow) * 16 + k]
                         : tc[(n * TJ + trow) * 16 + (k - 16)];
          e2  = (k < 16) ? tn[(n * TJ + trow) * 16 + k] : (unsigned short)0;
        }
        a01[2 * v + e] = e01; a2[2 * v + e] = e2;
      }
    }
    int o = lane & 15;
#pragma unroll
    for (int r = 0; r < 8; ++r) {
      int kb = b_kb(r, lane);
#pragma unroll
      for (int e = 0; e < 2; ++e) {
        int k = kb + e;
        float w01 = (k < 16) ? chw[(0 * 16 + k) * 16 + o]
                             : chw[(1 * 16 + (k - 16)) * 16 + o];
        float w2  = (k < 16) ? chw[(2 * 16 + k) * 16 + o] : 0.f;
        bw01[2 * r + e] = f2bf(w01);
        bw2[2 * r + e]  = f2bf(w2);
      }
    }
    v8f acc;
#pragma unroll
    for (int r = 0; r < 8; ++r) acc[r] = chb[o];
    acc = wmma_bf16(a01, bw01, acc);
    acc = wmma_bf16(a2,  bw2,  acc);
#pragma unroll
    for (int r = 0; r < 8; ++r) {
      int row = tb * 16 + c_row(r, lane);
      if (row < TJ)
        h2[(n * TJ + row) * 16 + col] = f2bf(fmaxf(acc[r], 0.f));
    }
  }
  __syncthreads();
}

// ---------------------------------------------------------------------------
// Temporal conv 16ch -> 32ch, kernel 3 (im2col K=48 padded to 64 -> 2 WMMAs
// per 16x16 tile, 2 output-channel tiles). Input LDS [32][TIN][16] ushort.
// ---------------------------------------------------------------------------
template <int TIN>
static __device__ __forceinline__ void tglu_wmma(
    const unsigned short* in, int n, int tbase,
    const float* __restrict__ w, const float* __restrict__ bias,
    int lane, v8f& accA, v8f& accG, int maxrow) {
  int trow = tbase + (lane & 15);
  v16us a0, a1;
#pragma unroll
  for (int v = 0; v < 8; ++v) {
    int kb = a_kb(v, lane);
#pragma unroll
    for (int e = 0; e < 2; ++e) {
      int kk = kb + e;                 // frag0 kappa 0..31, frag1 kappa 32..63
      int k0 = kk >> 4,        i0 = kk & 15;   // taps 0,1
      int k1 = (kk + 32) >> 4;                 // taps 2,3(pad)
      unsigned short e0 = 0, e1 = 0;
      if (trow < maxrow) {
        e0 = in[(n * TIN + (trow + k0)) * 16 + i0];
        e1 = (k1 < 3) ? in[(n * TIN + (trow + k1)) * 16 + i0] : (unsigned short)0;
      }
      a0[2 * v + e] = e0; a1[2 * v + e] = e1;
    }
  }
  int o = lane & 15;
#pragma unroll
  for (int ot = 0; ot < 2; ++ot) {
    v16us b0, b1;
    int og = ot * 16 + o;
#pragma unroll
    for (int r = 0; r < 8; ++r) {
      int kb = b_kb(r, lane);
#pragma unroll
      for (int e = 0; e < 2; ++e) {
        int kk = kb + e;
        int k0 = kk >> 4,        i0 = kk & 15;
        int k1 = (kk + 32) >> 4;
        b0[2 * r + e] = f2bf(w[(og * 16 + i0) * 3 + k0]);
        b1[2 * r + e] = (k1 < 3) ? f2bf(w[(og * 16 + i0) * 3 + k1]) : (unsigned short)0;
      }
    }
    v8f acc;
#pragma unroll
    for (int r = 0; r < 8; ++r) acc[r] = bias[og];
    acc = wmma_bf16(a0, b0, acc);
    acc = wmma_bf16(a1, b1, acc);
    if (ot == 0) accA = acc; else accG = acc;
  }
}

// ---------------------------------------------------------------------------
// Block 1: tglu1 -> cheb1 -> tglu2 (+res conv) -> pre-GN tensor + GN sums
// ---------------------------------------------------------------------------
struct Smem1 {
  float xs[N_ * TX];                       //  3360 B
  float red[512];                          //  2048 B
  unsigned short ltb[32 * 32];             //  2048 B
  unsigned short lt2b[32 * 32];            //  2048 B
  unsigned short h1[32 * TJ * 16];         // 26624 B each
  unsigned short tc[32 * TJ * 16];
  unsigned short tn[32 * TJ * 16];
  unsigned short h2[32 * TJ * 16];
};                                         // ~113 KB static LDS

__global__ __launch_bounds__(256) void block1_kernel(
    const float* __restrict__ x,     const float* __restrict__ lt,
    const float* __restrict__ t1w,   const float* __restrict__ t1b,
    const float* __restrict__ ch1w,  const float* __restrict__ ch1b,
    const float* __restrict__ t2w,   const float* __restrict__ t2b,
    const float* __restrict__ res1w, const float* __restrict__ res1b,
    float* __restrict__ hout, float* __restrict__ sums) {
  __shared__ Smem1 sm;
  const int b  = blockIdx.x / NTILES;
  const int t0 = (blockIdx.x % NTILES) * TT;
  const int tid = threadIdx.x;
  const int lane = tid & 31, wave = tid >> 5;

  for (int i = tid; i < 32 * 32; i += 256) {
    int r = i >> 5, c = i & 31;
    float v = (r < N_ && c < N_) ? lt[r * N_ + c] : 0.f;
    sm.ltb[i] = f2bf(v); sm.lt2b[i] = f2bf(2.f * v);
  }
  for (int i = tid; i < N_ * TX; i += 256) {           // x tile, halo 2, zero pad
    int n = i / TX, j = i % TX;
    int gt = t0 - 2 + j;
    float v = 0.f;
    if (gt >= 0 && gt < FT) {
      int e = gt / S_;
      v = x[(((size_t)b * E_ + e) * N_ + n) * S_ + (gt - e * S_)];
    }
    sm.xs[i] = v;
  }
  for (int i = tid; i < 2 * TJ * 16; i += 256) sm.h1[30 * TJ * 16 + i] = 0;
  __syncthreads();

  // tglu1: 1ch -> 16ch GLU (tiny K, VALU is right here)
  for (int i = tid; i < N_ * TJ; i += 256) {
    int n = i / TJ, j = i % TJ;
    float x0 = sm.xs[n * TX + j], x1 = sm.xs[n * TX + j + 1], x2 = sm.xs[n * TX + j + 2];
#pragma unroll
    for (int c = 0; c < HID; ++c) {
      float ga = t1b[c]       + t1w[c * 3] * x0 + t1w[c * 3 + 1] * x1 + t1w[c * 3 + 2] * x2;
      float gg = t1b[c + HID] + t1w[(c + HID) * 3] * x0 + t1w[(c + HID) * 3 + 1] * x1
                              + t1w[(c + HID) * 3 + 2] * x2;
      sm.h1[(n * TJ + j) * 16 + c] = f2bf(tanhf(ga) * sigm(gg));
    }
  }
  __syncthreads();

  cheb_stage(sm.h1, sm.tc, sm.tn, sm.h2, sm.ltb, sm.lt2b, ch1w, ch1b, lane, wave);

  // tglu2 + residual(1x1 conv of x) -> global pre-GN + sums
  float s1 = 0.f, s2 = 0.f;
  const int col = lane & 15;
  for (int task = wave; task < N_ * 2; task += 8) {
    int n = task >> 1, tb = task & 1;
    v8f accA, accG;
    tglu_wmma<TJ>(sm.h2, n, tb * 16, t2w, t2b, lane, accA, accG, TT);
#pragma unroll
    for (int r = 0; r < 8; ++r) {
      int trow = tb * 16 + c_row(r, lane);
      if (trow < TT) {
        float hval = tanhf(accA[r]) * sigm(accG[r]);
        float xv = sm.xs[n * TX + trow + 2];
        float out = hval + res1w[col] * xv + res1b[col];
        hout[(((size_t)b * N_ + n) * FT + (t0 + trow)) * HID + col] = out;
        s1 += out; s2 += out * out;
      }
    }
  }
  sm.red[tid] = s1; sm.red[256 + tid] = s2;
  __syncthreads();
  for (int st = 128; st > 0; st >>= 1) {
    if (tid < st) {
      sm.red[tid] += sm.red[tid + st];
      sm.red[256 + tid] += sm.red[256 + tid + st];
    }
    __syncthreads();
  }
  if (tid == 0) {
    atomicAdd(&sums[b * 2],     sm.red[0]);
    atomicAdd(&sums[b * 2 + 1], sm.red[256]);
  }
}

// ---------------------------------------------------------------------------
// Block 2: async-load raw -> gn1 -> tglu3 -> cheb2 -> tglu4 (+res) -> pre-GN2
// Raw f32 staging tile is union-aliased over h1/tc/tn (dead at load time).
// ---------------------------------------------------------------------------
struct Smem2 {
  unsigned short hin[32 * TX * 16];        // 28672 B
  float red[512];                          //  2048 B
  unsigned short ltb[32 * 32];
  unsigned short lt2b[32 * 32];
  union {
    float raw[N_ * TX * 16];               // 53760 B (async staging)
    struct {
      unsigned short h1[32 * TJ * 16];
      unsigned short tc[32 * TJ * 16];
      unsigned short tn[32 * TJ * 16];
    } c;                                   // 79872 B
  } u;
  unsigned short h2[32 * TJ * 16];
};                                         // ~138 KB static LDS

__global__ __launch_bounds__(256) void block2_kernel(
    const float* __restrict__ hin_g, const float* __restrict__ stats,
    const float* __restrict__ lt,
    const float* __restrict__ g1,   const float* __restrict__ b1g,
    const float* __restrict__ t3w,  const float* __restrict__ t3b,
    const float* __restrict__ ch2w, const float* __restrict__ ch2b,
    const float* __restrict__ t4w,  const float* __restrict__ t4b,
    float* __restrict__ hout, float* __restrict__ sums) {
  __shared__ Smem2 sm;
  const int b  = blockIdx.x / NTILES;
  const int t0 = (blockIdx.x % NTILES) * TT;
  const int tid = threadIdx.x;
  const int lane = tid & 31, wave = tid >> 5;
  const float mean = stats[b * 2], rstd = stats[b * 2 + 1];

  // Async bulk copy: raw f32 tile (gfx1250 global->LDS async, ASYNCcnt tracked).
  // Per timestep: 16ch * 4B = 64B = 4 x b128 chunks.
  const int CHN = TX * 4;                  // chunks per node
  for (int idx = tid; idx < N_ * CHN; idx += 256) {
    int n = idx / CHN, c = idx % CHN;
    int j = c >> 2;
    int gt = t0 - 2 + j;
    if (gt >= 0 && gt < FT) {
      unsigned long long g = (unsigned long long)(const void*)
          (hin_g + ((size_t)(b * N_ + n) * FT + gt) * HID + (c & 3) * 4);
      unsigned ldsb = (unsigned)(offsetof(Smem2, u)
                     + ((n * TX + j) * 16 + (c & 3) * 4) * sizeof(float));
      asm volatile("global_load_async_to_lds_b128 %0, %1, off"
                   :: "v"(ldsb), "v"(g) : "memory");
    }
  }
  for (int i = tid; i < 32 * 32; i += 256) {
    int r = i >> 5, c = i & 31;
    float v = (r < N_ && c < N_) ? lt[r * N_ + c] : 0.f;
    sm.ltb[i] = f2bf(v); sm.lt2b[i] = f2bf(2.f * v);
  }
  asm volatile("s_wait_asynccnt 0x0" ::: "memory");
  __syncthreads();

  // Apply GN1 affine while compressing f32 -> bf16 into hin.
  for (int i = tid; i < N_ * TX * HID; i += 256) {
    int c = i % HID; int j = (i / HID) % TX; int n = i / (HID * TX);
    int gt = t0 - 2 + j;
    float v = 0.f;
    if (gt >= 0 && gt < FT)
      v = (sm.u.raw[(n * TX + j) * 16 + c] - mean) * rstd * g1[c] + b1g[c];
    sm.hin[(n * TX + j) * 16 + c] = f2bf(v);
  }
  for (int i = tid; i < 2 * TX * 16; i += 256) sm.hin[30 * TX * 16 + i] = 0;
  __syncthreads();                          // raw dead beyond this point

  // tglu3 (WMMA) + zero node-pad rows of h1 (aliases raw, now free)
  for (int i = tid; i < 2 * TJ * 16; i += 256) sm.u.c.h1[30 * TJ * 16 + i] = 0;
  for (int task = wave; task < N_ * 2; task += 8) {
    int n = task >> 1, tb = task & 1;
    v8f accA, accG;
    tglu_wmma<TX>(sm.hin, n, tb * 16, t3w, t3b, lane, accA, accG, TJ);
    const int col_ = lane & 15;
#pragma unroll
    for (int r = 0; r < 8; ++r) {
      int trow = tb * 16 + c_row(r, lane);
      if (trow < TJ)
        sm.u.c.h1[(n * TJ + trow) * 16 + col_] = f2bf(tanhf(accA[r]) * sigm(accG[r]));
    }
  }
  __syncthreads();

  cheb_stage(sm.u.c.h1, sm.u.c.tc, sm.u.c.tn, sm.h2, sm.ltb, sm.lt2b,
             ch2w, ch2b, lane, wave);

  float s1 = 0.f, s2 = 0.f;
  const int col = lane & 15;
  for (int task = wave; task < N_ * 2; task += 8) {
    int n = task >> 1, tb = task & 1;
    v8f accA, accG;
    tglu_wmma<TJ>(sm.h2, n, tb * 16, t4w, t4b, lane, accA, accG, TT);
#pragma unroll
    for (int r = 0; r < 8; ++r) {
      int trow = tb * 16 + c_row(r, lane);
      if (trow < TT) {
        float hval = tanhf(accA[r]) * sigm(accG[r]);
        float out = hval + bf2f(sm.hin[(n * TX + trow + 2) * 16 + col]);
        hout[(((size_t)b * N_ + n) * FT + (t0 + trow)) * HID + col] = out;
        s1 += out; s2 += out * out;
      }
    }
  }
  sm.red[tid] = s1; sm.red[256 + tid] = s2;
  __syncthreads();
  for (int st = 128; st > 0; st >>= 1) {
    if (tid < st) {
      sm.red[tid] += sm.red[tid + st];
      sm.red[256 + tid] += sm.red[256 + tid + st];
    }
    __syncthreads();
  }
  if (tid == 0) {
    atomicAdd(&sums[b * 2],     sm.red[0]);
    atomicAdd(&sums[b * 2 + 1], sm.red[256]);
  }
}

__global__ void zero_kernel(float* __restrict__ p, int n) {
  int i = blockIdx.x * blockDim.x + threadIdx.x;
  if (i < n) p[i] = 0.f;
}

__global__ void stats_kernel(const float* __restrict__ sums, float* __restrict__ stats) {
  int b = threadIdx.x;
  if (b < B_) {
    const float cnt = (float)((size_t)HID * N_ * FT);
    float m = sums[b * 2] / cnt;
    float var = sums[b * 2 + 1] / cnt - m * m;
    stats[b * 2] = m;
    stats[b * 2 + 1] = rsqrtf(var + EPSV);
  }
}

__global__ __launch_bounds__(256) void head_kernel(
    const float* __restrict__ hb, const float* __restrict__ stats,
    const float* __restrict__ g2, const float* __restrict__ b2,
    const float* __restrict__ hw, const float* __restrict__ hbias,
    float* __restrict__ out) {
  int idx = blockIdx.x * 256 + threadIdx.x;           // over B*N*S
  if (idx >= B_ * N_ * S_) return;
  int s = idx % S_; int n = (idx / S_) % N_; int b = idx / (S_ * N_);
  float mean = stats[b * 2], rstd = stats[b * 2 + 1];
  const float* p = &hb[(((size_t)b * N_ + n) * FT + (S_ + s)) * HID]; // center slice
  float acc = hbias[0];
#pragma unroll
  for (int c = 0; c < HID; ++c)
    acc += ((p[c] - mean) * rstd * g2[c] + b2[c]) * hw[c];
  out[idx] = acc;
}

extern "C" void kernel_launch(void* const* d_in, const int* in_sizes, int n_in,
                              void* d_out, int out_size, void* d_ws, size_t ws_size,
                              hipStream_t stream) {
  const float* x     = (const float*)d_in[0];
  const float* lt    = (const float*)d_in[1];
  const float* t1w   = (const float*)d_in[2];
  const float* t1b   = (const float*)d_in[3];
  const float* ch1w  = (const float*)d_in[4];
  const float* ch1b  = (const float*)d_in[5];
  const float* t2w   = (const float*)d_in[6];
  const float* t2b   = (const float*)d_in[7];
  const float* res1w = (const float*)d_in[8];
  const float* res1b = (const float*)d_in[9];
  const float* gn1g  = (const float*)d_in[10];
  const float* gn1b  = (const float*)d_in[11];
  const float* t3w   = (const float*)d_in[12];
  const float* t3b   = (const float*)d_in[13];
  const float* ch2w  = (const float*)d_in[14];
  const float* ch2b  = (const float*)d_in[15];
  const float* t4w   = (const float*)d_in[16];
  const float* t4b   = (const float*)d_in[17];
  const float* gn2g  = (const float*)d_in[18];
  const float* gn2b  = (const float*)d_in[19];
  const float* hw    = (const float*)d_in[20];
  const float* hb    = (const float*)d_in[21];

  const size_t HA = (size_t)B_ * N_ * FT * HID;       // 94.4M f32 per pre-GN tensor
  float* h_a    = (float*)d_ws;
  float* h_b    = h_a + HA;
  float* sums1  = h_b + HA;
  float* sums2  = sums1 + 2 * B_;
  float* stats1 = sums2 + 2 * B_;
  float* stats2 = stats1 + 2 * B_;

  zero_kernel<<<1, 128, 0, stream>>>(sums1, 4 * B_);

  dim3 grid(B_ * NTILES);
  block1_kernel<<<grid, 256, 0, stream>>>(x, lt, t1w, t1b, ch1w, ch1b,
                                          t2w, t2b, res1w, res1b, h_a, sums1);
  stats_kernel<<<1, 64, 0, stream>>>(sums1, stats1);
  block2_kernel<<<grid, 256, 0, stream>>>(h_a, stats1, lt, gn1g, gn1b,
                                          t3w, t3b, ch2w, ch2b, t4w, t4b,
                                          h_b, sums2);
  stats_kernel<<<1, 64, 0, stream>>>(sums2, stats2);

  int tot = B_ * N_ * S_;
  head_kernel<<<(tot + 255) / 256, 256, 0, stream>>>(h_b, stats2, gn2g, gn2b,
                                                     hw, hb, (float*)d_out);
}
